// VectorQuantizer_14474039788284
// MI455X (gfx1250) — matrix operand order repaired and checked by reading, not compiled, not run on hardware
//
#include <hip/hip_runtime.h>

typedef __attribute__((ext_vector_type(16))) _Float16     v16h;
typedef __attribute__((ext_vector_type(8)))  float        v8f;
typedef __attribute__((ext_vector_type(4)))  unsigned int u32x4;
typedef __attribute__((ext_vector_type(8)))  int          i32x8;
typedef __attribute__((ext_vector_type(4)))  int          i32x4;

#define DDIM  64
#define CHUNK 256   // codes staged per TDM transfer: 256*64*2B = 32 KB

// ---------------------------------------------------------------------------
// TDM: 1-D tensor DMA of nelem f16 elements from global -> LDS.
// D# built per CDNA5 ISA ch.8: group0 = {count=1, lds_addr, global_addr,
// type=2}; group1 = {data_size=2B, tensor_dim0=tile_dim0=nelem, stride=nelem}.
// Issued once per wave (EXEC ignored); completion tracked via TENSORcnt.
// ---------------------------------------------------------------------------
__device__ __forceinline__ void tdm_load_1d(unsigned lds_byte_off,
                                            const void* gptr,
                                            unsigned nelem_f16) {
  unsigned long long ga = (unsigned long long)(uintptr_t)gptr;
  u32x4 g0;
  g0[0] = 1u;                                   // count=1, user mode
  g0[1] = lds_byte_off;                         // lds_addr
  g0[2] = (unsigned)(ga & 0xFFFFFFFFu);         // global_addr[31:0]
  g0[3] = (unsigned)((ga >> 32) & 0x1FFFFFFu)   // global_addr[56:32]
          | (2u << 30);                         // type = 2 ("image")
  i32x8 g1;
  g1[0] = (int)(1u << 16);                      // mask=0, data_size=1 (2 B)
  g1[1] = (int)((nelem_f16 & 0xFFFFu) << 16);   // tensor_dim0[15:0]
  g1[2] = (int)(((nelem_f16 >> 16) & 0xFFFFu)   // tensor_dim0[31:16]
          | (1u << 16));                        // tensor_dim1 = 1
  g1[3] = (int)((nelem_f16 & 0xFFFFu) << 16);   // tile_dim0 = nelem
  g1[4] = 0;                                    // tile_dim1/2 unused
  g1[5] = (int)nelem_f16;                       // tensor_dim0_stride[31:0]
  g1[6] = 0;
  g1[7] = 0;
  i32x4 z4 = {0, 0, 0, 0};
#if __clang_major__ >= 23
  i32x8 z8 = {0, 0, 0, 0, 0, 0, 0, 0};
  __builtin_amdgcn_tensor_load_to_lds(g0, g1, z4, z4, z8, 0);
#else
  __builtin_amdgcn_tensor_load_to_lds(g0, g1, z4, z4, 0);
#endif
}

// LDS byte offset of a __shared__ object (generic LDS addr low 32 bits).
__device__ __forceinline__ unsigned lds_offset_of(const void* p) {
  return (unsigned)(uintptr_t)p;
}

// ---------------------------------------------------------------------------
// Kernel 1: codebook f32 -> f16 (WMMA operand precision) + ||c||^2
// ---------------------------------------------------------------------------
__global__ __launch_bounds__(256) void vq_prep_kernel(
    const float* __restrict__ codebook, _Float16* __restrict__ cbf16,
    float* __restrict__ cnorm, int K) {
  int c = blockIdx.x * blockDim.x + threadIdx.x;
  if (c >= K) return;
  const float* row = codebook + (size_t)c * DDIM;
  _Float16* dst = cbf16 + (size_t)c * DDIM;
  float n = 0.f;
#pragma unroll
  for (int d = 0; d < DDIM; ++d) {
    float v = row[d];
    n += v * v;
    dst[d] = (_Float16)v;
  }
  cnorm[c] = n;
}

// ---------------------------------------------------------------------------
// Kernel 2: TDM-staged, WMMA-based nearest-code search with fused argmin,
// gather, straight-through output, per-block loss partials.
// 256 threads = 8 wave32 waves; each wave owns TWO 16-row groups (32 rows)
// so every B tile staged in LDS feeds 4 WMMAs.
// ---------------------------------------------------------------------------
__global__ __launch_bounds__(256) void vq_main_kernel(
    const float* __restrict__ x, const float* __restrict__ y,
    const float* __restrict__ codebook, const _Float16* __restrict__ cbf16,
    const float* __restrict__ cnorm, float* __restrict__ out,
    float* __restrict__ partials, int K) {
  __shared__ __align__(32) _Float16 sB[2][CHUNK * DDIM];  // 2 x 32 KB
  __shared__ int   s_idx[8][32];
  __shared__ float s_part[8][2];

  const int lane = threadIdx.x & 31;
  const int wave = threadIdx.x >> 5;
  const int col  = lane & 15;   // WMMA C column (N) held by this lane
  const int hi   = lane >> 4;   // 0: lanes 0-15, 1: lanes 16-31
  const size_t rowBase = (size_t)blockIdx.x * 256 + (size_t)wave * 32;

  // ---- A operands: two 16x64 row groups of x, f32->f16, contiguous k layout:
  // lanes 0-15 hold M=lane, K=0..15; lanes 16-31 hold K=16..31 (matches B)
  const float* xr0 = x + (rowBase + col) * DDIM;
  const float* xr1 = x + (rowBase + 16 + col) * DDIM;
  const int aoff = hi * 16;
  v16h a00, a01, a10, a11;
#pragma unroll
  for (int i = 0; i < 8; ++i) {
    a00[i]     = (_Float16)xr0[aoff + i];
    a00[8 + i] = (_Float16)xr0[8 + aoff + i];
    a01[i]     = (_Float16)xr0[32 + aoff + i];
    a01[8 + i] = (_Float16)xr0[40 + aoff + i];
    a10[i]     = (_Float16)xr1[aoff + i];
    a10[8 + i] = (_Float16)xr1[8 + aoff + i];
    a11[i]     = (_Float16)xr1[32 + aoff + i];
    a11[8 + i] = (_Float16)xr1[40 + aoff + i];
  }

  float best0[8], best1[8];
  int   bidx0[8], bidx1[8];
#pragma unroll
  for (int v = 0; v < 8; ++v) {
    best0[v] = 3.4e38f; bidx0[v] = 0;
    best1[v] = 3.4e38f; bidx1[v] = 0;
  }

  const int koff = hi * 16;  // B layout: lanes 0-15 hold K 0-15, 16-31 hold 16-31
  const int nchunks = K / CHUNK;

  // prime the double buffer: DMA chunk 0
  if (wave == 0) tdm_load_1d(lds_offset_of(&sB[0][0]), cbf16, CHUNK * DDIM);

  int cur = 0;
  for (int c = 0; c < nchunks; ++c) {
    if (wave == 0) __builtin_amdgcn_s_wait_tensorcnt(0);  // chunk c landed
    __syncthreads();                                      // publish to all waves
    if (wave == 0 && c + 1 < nchunks)                     // overlap next DMA
      tdm_load_1d(lds_offset_of(&sB[cur ^ 1][0]),
                  cbf16 + (size_t)(c + 1) * CHUNK * DDIM, CHUNK * DDIM);

    const int cbase = c * CHUNK;
    __builtin_prefetch(cnorm + cbase + CHUNK, 0, 1);      // next chunk's norms

    for (int tt = 0; tt < CHUNK / 16; ++tt) {
      const int code = cbase + tt * 16 + col;
      const _Float16* cb = &sB[cur][(tt * 16 + col) * DDIM + koff];
      v16h b0 = *(const v16h*)(cb);        // d 0..31 half for this lane group
      v16h b1 = *(const v16h*)(cb + 32);   // d 32..63 half
      float cn = cnorm[code];

      v8f acc0 = {}, acc1 = {};
      acc0 = __builtin_amdgcn_wmma_f32_16x16x32_f16(false, a00, false, b0,
                                                    (short)0, acc0, false, false);
      acc1 = __builtin_amdgcn_wmma_f32_16x16x32_f16(false, a10, false, b0,
                                                    (short)0, acc1, false, false);
      acc0 = __builtin_amdgcn_wmma_f32_16x16x32_f16(false, a01, false, b1,
                                                    (short)0, acc0, false, false);
      acc1 = __builtin_amdgcn_wmma_f32_16x16x32_f16(false, a11, false, b1,
                                                    (short)0, acc1, false, false);

      // score(row, code) = ||c||^2 - 2*x.c ; idx monotone -> first-min kept
#pragma unroll
      for (int v = 0; v < 8; ++v) {
        float s0 = cn - 2.0f * acc0[v];
        float s1 = cn - 2.0f * acc1[v];
        if (s0 < best0[v]) { best0[v] = s0; bidx0[v] = code; }
        if (s1 < best1[v]) { best1[v] = s1; bidx1[v] = code; }
      }
    }
    __syncthreads();   // all waves done reading sB[cur] before it is refilled
    cur ^= 1;
  }

  // ---- argmin across the 16 lanes that share each row (C layout: lane%16=N)
#pragma unroll
  for (int m = 8; m >= 1; m >>= 1) {
#pragma unroll
    for (int v = 0; v < 8; ++v) {
      float os0 = __shfl_xor(best0[v], m, 16);
      int   oi0 = __shfl_xor(bidx0[v], m, 16);
      if (os0 < best0[v] || (os0 == best0[v] && oi0 < bidx0[v])) {
        best0[v] = os0; bidx0[v] = oi0;
      }
      float os1 = __shfl_xor(best1[v], m, 16);
      int   oi1 = __shfl_xor(bidx1[v], m, 16);
      if (os1 < best1[v] || (os1 == best1[v] && oi1 < bidx1[v])) {
        best1[v] = os1; bidx1[v] = oi1;
      }
    }
  }
  if (col == 0) {
#pragma unroll
    for (int v = 0; v < 8; ++v) {
      s_idx[wave][hi * 8 + v]      = bidx0[v];   // rows 0..15 of this wave
      s_idx[wave][16 + hi * 8 + v] = bidx1[v];   // rows 16..31
    }
  }
  __syncthreads();

  // ---- gather winning codes, straight-through output, loss partials ----
  float s1 = 0.f, s2 = 0.f;
  for (int r = 0; r < 32; ++r) {
    const int idx = s_idx[wave][r];
    const size_t ro = (rowBase + (size_t)r) * DDIM;
    const float* q = codebook + (size_t)idx * DDIM;
#pragma unroll
    for (int dd = 0; dd < 2; ++dd) {
      const int d = lane + dd * 32;
      float qv = q[d], xv = x[ro + d], yv = y[ro + d];
      float t0 = xv + yv;
      out[ro + d] = t0 + (qv - t0);   // exact reference rounding of STE output
      float d1 = qv - xv, d2 = qv - yv;
      s1 += d1 * d1;
      s2 += d2 * d2;
    }
  }
#pragma unroll
  for (int m = 16; m >= 1; m >>= 1) {
    s1 += __shfl_xor(s1, m, 32);
    s2 += __shfl_xor(s2, m, 32);
  }
  if (lane == 0) { s_part[wave][0] = s1; s_part[wave][1] = s2; }
  __syncthreads();
  if (threadIdx.x == 0) {
    float t1 = 0.f, t2 = 0.f;
    for (int w = 0; w < 8; ++w) { t1 += s_part[w][0]; t2 += s_part[w][1]; }
    partials[(size_t)blockIdx.x * 2]     = t1;
    partials[(size_t)blockIdx.x * 2 + 1] = t2;
  }
}

// ---------------------------------------------------------------------------
// Kernel 3: deterministic fixed-order reduction of block partials -> loss
// ---------------------------------------------------------------------------
__global__ __launch_bounds__(256) void vq_finish_kernel(
    const float* __restrict__ partials, float* __restrict__ loss_out,
    int nblocks, float inv_nd) {
  __shared__ float sh[256][2];
  float t1 = 0.f, t2 = 0.f;
  for (int b = threadIdx.x; b < nblocks; b += 256) {
    t1 += partials[2 * b];
    t2 += partials[2 * b + 1];
  }
  sh[threadIdx.x][0] = t1;
  sh[threadIdx.x][1] = t2;
  __syncthreads();
  for (int s = 128; s >= 1; s >>= 1) {
    if (threadIdx.x < s) {
      sh[threadIdx.x][0] += sh[threadIdx.x + s][0];
      sh[threadIdx.x][1] += sh[threadIdx.x + s][1];
    }
    __syncthreads();
  }
  if (threadIdx.x == 0) {
    // loss = q + qs + 0.25*(e + s) = 1.25*(mean((q-x)^2) + mean((q-y)^2))
    *loss_out = 1.25f * (sh[0][0] + sh[0][1]) * inv_nd;
  }
}

// ---------------------------------------------------------------------------
extern "C" void kernel_launch(void* const* d_in, const int* in_sizes, int n_in,
                              void* d_out, int out_size, void* d_ws, size_t ws_size,
                              hipStream_t stream) {
  const float* x        = (const float*)d_in[0];
  const float* y        = (const float*)d_in[1];
  const float* codebook = (const float*)d_in[2];
  float* out = (float*)d_out;

  const int D = DDIM;
  const int N = in_sizes[0] / D;   // 131072
  const int K = in_sizes[2] / D;   // 4096

  char* ws = (char*)d_ws;
  _Float16* cbf16 = (_Float16*)ws;                                   // 512 KB
  float* cnorm    = (float*)(ws + (size_t)K * D * sizeof(_Float16)); //  16 KB
  float* partials = (float*)(ws + (size_t)K * D * sizeof(_Float16)
                                + (size_t)K * sizeof(float));        //   4 KB

  const int nblocks = N / 256;     // 512 blocks, 8 waves x 32 rows each
  vq_prep_kernel<<<(K + 255) / 256, 256, 0, stream>>>(codebook, cbf16, cnorm, K);
  vq_main_kernel<<<nblocks, 256, 0, stream>>>(x, y, codebook, cbf16, cnorm,
                                              out, partials, K);
  vq_finish_kernel<<<1, 256, 0, stream>>>(partials, out + (size_t)N * D,
                                          nblocks, 1.0f / (float)((size_t)N * D));
}
